// VectorQuantizerEMA_343597384065
// MI455X (gfx1250) — matrix operand order repaired and checked by reading, not compiled, not run on hardware
//
#include <hip/hip_runtime.h>

typedef __attribute__((ext_vector_type(16))) _Float16 v16h;
typedef __attribute__((ext_vector_type(8)))  _Float16 v8h;
typedef __attribute__((ext_vector_type(8)))  float    v8f;
typedef __attribute__((ext_vector_type(4)))  float    f4;

#define K_CODES 400
#define K_PAD   416     // padded to 26 whole 16-code tiles
#define DIM     64
#define NTILES  26
#define LSTR    72      // padded LDS row stride (halves): 144B, 16B-aligned, bank-conflict-free
#define ITERS   4       // 16-row tiles per wave per block

union V16U { v16h v; v8h h[2]; };

__global__ void __launch_bounds__(256)
vq_wmma_kernel(const float* __restrict__ x,
               const float* __restrict__ emb,
               float* __restrict__ out)
{
    __shared__ _Float16 sEmb[K_PAD * LSTR];   // 59904 B
    __shared__ float    sNorm[K_PAD];         // 1664 B  (holds -0.5*||e||^2)

    const int tid = threadIdx.x;

    // Stage codebook f32 -> f16 into LDS; pad tiles 400..415 with zeros.
    for (int i = tid; i < K_PAD * DIM; i += 256) {
        int c = i >> 6, d = i & 63;
        float v = (c < K_CODES) ? emb[i] : 0.0f;
        sEmb[c * LSTR + d] = (_Float16)v;
    }
    // argmin_k(||e||^2 - 2 x.e)  ==  argmax_k(x.e - 0.5*||e||^2)
    // Stage -0.5*||e||^2 so it can seed the WMMA accumulator directly.
    // Padded codes get -3e38 so they can never win the argmax.
    for (int c = tid; c < K_PAD; c += 256) {
        float s = -3.0e38f;
        if (c < K_CODES) {
            const float* e = emb + c * DIM;
            float acc = 0.0f;
            #pragma unroll
            for (int d = 0; d < DIM; ++d) { float v = e[d]; acc += v * v; }
            s = -0.5f * acc;
        }
        sNorm[c] = s;
    }
    __syncthreads();

    const int lane = tid & 31;
    const int wave = tid >> 5;
    const int hi   = lane >> 4;   // half-wave (K-group select)
    const int lm   = lane & 15;   // M (A rows) / N (B cols) within tile

    for (int it = 0; it < ITERS; ++it) {
        const long  rowBase = ((long)blockIdx.x * ITERS + it) * 128 + wave * 16;
        const float* xrow   = x + (rowBase + lm) * DIM;

        // Prefetch next iteration's rows while we compute this one.
        if (it + 1 < ITERS) __builtin_prefetch(xrow + 128 * DIM, 0, 1);

        // A fragments: 16x32 f16 per WMMA. ISA layout:
        //   lanes 0-15: VGPR0-3 = K 0..7, VGPR4-7 = K 16..23
        //   lanes16-31: VGPR0-3 = K 8..15, VGPR4-7 = K 24..31
        V16U a0, a1;
        #pragma unroll
        for (int h = 0; h < 8; ++h) {
            a0.v[h]     = (_Float16)xrow[hi * 8 + h];
            a0.v[h + 8] = (_Float16)xrow[16 + hi * 8 + h];
            a1.v[h]     = (_Float16)xrow[32 + hi * 8 + h];
            a1.v[h + 8] = (_Float16)xrow[48 + hi * 8 + h];
        }

        float best[8];
        int   midx[8];
        #pragma unroll
        for (int j = 0; j < 8; ++j) { best[j] = -3.3e38f; midx[j] = 0; }

        // Two code tiles per trip; ROLLED (unroll 1) — full unroll spills.
        // Independent accumulators hide the WMMA->VALU hazard window.
        #pragma unroll 1
        for (int t = 0; t < NTILES; t += 2) {
            const int code0 = t * 16 + lm;
            const int code1 = code0 + 16;
            const _Float16* bp0 = &sEmb[code0 * LSTR + hi * 8];
            const _Float16* bp1 = bp0 + 16 * LSTR;

            V16U b00, b01, b10, b11;
            b00.h[0] = *(const v8h*)(bp0);
            b00.h[1] = *(const v8h*)(bp0 + 16);
            b01.h[0] = *(const v8h*)(bp0 + 32);
            b01.h[1] = *(const v8h*)(bp0 + 48);
            b10.h[0] = *(const v8h*)(bp1);
            b10.h[1] = *(const v8h*)(bp1 + 16);
            b11.h[0] = *(const v8h*)(bp1 + 32);
            b11.h[1] = *(const v8h*)(bp1 + 48);

            // Seed C with -0.5*||e||^2 (same column for every slot of a lane):
            // the WMMA emits the score directly; splat movs sit BEFORE the
            // WMMA where they dual-issue into idle slots, not after it.
            const float mn0 = sNorm[code0];
            const float mn1 = sNorm[code1];
            v8f acc0 = {mn0, mn0, mn0, mn0, mn0, mn0, mn0, mn0};
            v8f acc1 = {mn1, mn1, mn1, mn1, mn1, mn1, mn1, mn1};

            acc0 = __builtin_amdgcn_wmma_f32_16x16x32_f16(false, a0.v, false, b00.v,
                                                          (short)0, acc0, false, false);
            acc0 = __builtin_amdgcn_wmma_f32_16x16x32_f16(false, a1.v, false, b01.v,
                                                          (short)0, acc0, false, false);
            acc1 = __builtin_amdgcn_wmma_f32_16x16x32_f16(false, a0.v, false, b10.v,
                                                          (short)0, acc1, false, false);
            acc1 = __builtin_amdgcn_wmma_f32_16x16x32_f16(false, a1.v, false, b11.v,
                                                          (short)0, acc1, false, false);

            // acc VGPR j, this lane: row = j + 8*hi, col = lm.
            #pragma unroll
            for (int j = 0; j < 8; ++j) {
                bool t0 = acc0[j] > best[j];
                midx[j] = t0 ? code0   : midx[j];
                best[j] = t0 ? acc0[j] : best[j];
                bool t1 = acc1[j] > best[j];
                midx[j] = t1 ? code1   : midx[j];
                best[j] = t1 ? acc1[j] : best[j];
            }
        }

        // Branchless argmax butterfly across the 16 lanes of each half-wave
        // (ties resolve to the lower code index, matching jnp.argmin).
        #pragma unroll
        for (int j = 0; j < 8; ++j) {
            #pragma unroll
            for (int m = 1; m < 16; m <<= 1) {
                float ob = __shfl_xor(best[j], m, 32);
                int   oi = __shfl_xor(midx[j], m, 32);
                bool  take = (ob > best[j]) | ((ob == best[j]) & (oi < midx[j]));
                best[j] = take ? ob : best[j];
                midx[j] = take ? oi : midx[j];
            }
        }

        // Gather winning fp32 code rows to output: 16 lanes x float4 = 64 floats/row.
        #pragma unroll
        for (int j = 0; j < 8; ++j) {
            const long row = rowBase + j + hi * 8;
            const f4* src = (const f4*)(emb + (long)midx[j] * DIM);
            f4*       dst = (f4*)(out + row * DIM);
            dst[lm] = src[lm];
        }
    }
}

extern "C" void kernel_launch(void* const* d_in, const int* in_sizes, int n_in,
                              void* d_out, int out_size, void* d_ws, size_t ws_size,
                              hipStream_t stream)
{
    const float* x   = (const float*)d_in[0];
    const float* emb = (const float*)d_in[1];
    float*       out = (float*)d_out;

    const int rows = in_sizes[0] / DIM;          // 262144
    const int grid = rows / (128 * ITERS);       // 512 blocks x 256 threads (8 waves)
    vq_wmma_kernel<<<grid, 256, 0, stream>>>(x, emb, out);
}
